// TopLoss_34591666602711
// MI455X (gfx1250) — compile-verified, matrix operand order checked by reading, not run on hardware
//
#include <hip/hip_runtime.h>
#include <hip/hip_bf16.h>

// ---------------------------------------------------------------------------
// TopLoss (0-dim + 1-dim superlevel persistence) on a 256x256 float32 image.
// Pipeline: build stable sort keys -> bitonic sort (LDS-fused inner stages,
//           TDM global->LDS staging) -> serial elder-rule union-find with
//           parent[] in LDS -> WMMA (f32_16x16x32_f16 hi/lo split) reduction
//           of bar lengths -> finalize (sum partials, subtract longest H1 bar).
// ---------------------------------------------------------------------------

typedef __attribute__((ext_vector_type(16))) _Float16 v16h;
typedef __attribute__((ext_vector_type(8)))  float    v8f;
typedef unsigned int u32x4 __attribute__((ext_vector_type(4)));
typedef int          i32x4 __attribute__((ext_vector_type(4)));
typedef int          i32x8 __attribute__((ext_vector_type(8)));

#define NVERT     65536      // 256*256
#define N_EDGE_A  195585     // 65280 + 65280 + 65025
#define N_EDGE_B  196605     // 195585 + 1020 boundary edges
#define NPAD      262144     // 2^18, bitonic size
#define LEN_CAP   131072     // 2*65536 length slots (zero padded)
#define SLOC      4096       // u64 elements per LDS sort chunk (32 KB)

// --------------------------- TDM staging helper ----------------------------
// DMA one contiguous 32KB chunk (4096 x u64) from global to LDS via the
// Tensor Data Mover. D# group0: count=1, lds_addr, global_addr, type=2.
// group1: data_size=8B, tensor_dim0=tile_dim0=4096, stride=4096, no
// mask/pad/iterate. Arity differs across toolchains (5-arg ROCm 7.2 builtin
// vs 6-arg therock headers) -> discriminate via the TDM header's presence.
// Descriptor field semantics are not hardware-verified, so callers follow up
// with an authoritative VALU overwrite of the same LDS range after a barrier.
__device__ __forceinline__ void tdm_load_chunk(const void* gsrc, unsigned lds_off) {
#if __has_builtin(__builtin_amdgcn_tensor_load_to_lds)
    unsigned long long ga = (unsigned long long)(uintptr_t)gsrc;
    u32x4 g0;
    g0[0] = 1u;                                                   // count=1
    g0[1] = lds_off;                                              // lds_addr
    g0[2] = (unsigned)(ga & 0xFFFFFFFFull);                       // gaddr[31:0]
    g0[3] = (unsigned)((ga >> 32) & 0x01FFFFFFull) | (2u << 30);  // gaddr[56:32], type=2
    i32x8 g1;
    g1[0] = (int)(3u << 16);        // data_size = 8 bytes; mask/flags = 0
    g1[1] = (int)(4096u << 16);     // tensor_dim0[15:0] (bits 79:48)
    g1[2] = (int)(1u << 16);        // tensor_dim1 = 1
    g1[3] = (int)(4096u << 16);     // tile_dim0 = 4096
    g1[4] = 0;                      // tile_dim1/2 unused (1D tile)
    g1[5] = 4096;                   // tensor_dim0_stride[31:0]
    g1[6] = 0;
    g1[7] = 0;
    i32x4 z4 = {0, 0, 0, 0};
#if __has_include(<hip/amd_detail/amd_gfx1250_TDM.h>)
    i32x8 z8 = {0, 0, 0, 0, 0, 0, 0, 0};
    __builtin_amdgcn_tensor_load_to_lds(g0, g1, z4, z4, z8, 0);
#else
    __builtin_amdgcn_tensor_load_to_lds(g0, g1, z4, z4, 0);
#endif
#if __has_builtin(__builtin_amdgcn_s_wait_tensorcnt)
    __builtin_amdgcn_s_wait_tensorcnt(0);
#endif
#else
    (void)gsrc; (void)lds_off;
#endif
}

// ----------------------------- edge topology -------------------------------
// Sorted-unique boundary vertex by rank (matches np.unique order).
__device__ __forceinline__ int bvert(int j) {
    if (j < 256) return j;                               // row 0
    if (j < 764) {                                       // rows 1..254, cols {0,255}
        int k = j - 256;
        int row = 1 + (k >> 1);
        int col = (k & 1) ? 255 : 0;
        return row * 256 + col;
    }
    return 65280 + (j - 764);                            // row 255
}

// Edge decode: [horizontal | vertical | diagonal | (B only) boundary]
__device__ __forceinline__ void edge_uv(int i, int sel, int& u, int& v) {
    if (i < 65280) {                      // horizontal: idx[:, :-1] -> idx[:, 1:]
        int r = i / 255, c = i - r * 255;
        u = r * 256 + c; v = u + 1;
    } else if (i < 130560) {              // vertical: idx[:-1, :] -> idx[1:, :]
        u = i - 65280; v = u + 256;
    } else if (i < 195585) {              // diagonal (255x255 cells)
        int j = i - 130560;
        int r = j / 255, c = j - r * 255;
        if (!sel) { u = r * 256 + c;       v = u + 257; }          // main diag
        else      { u = (r + 1) * 256 + c; v = r * 256 + c + 1; }  // anti diag
    } else {                              // boundary -> virtual node (sel==1 only)
        u = bvert(i - 195585); v = NVERT;
    }
}

__device__ __forceinline__ float gval(const float* f, int x, int sel) {
    if (sel) return (x == NVERT) ? -INFINITY : f[x];  // sublevel of f (+virtual -inf)
    return -f[x];                                     // sublevel of -f (superlevel of f)
}

// Order-preserving float -> u32 key. Canonicalize -0.0 so numpy tie semantics hold.
__device__ __forceinline__ unsigned int ordkey(float x) {
    unsigned int b = __float_as_uint(x);
    if ((b << 1) == 0u) b = 0u;
    return (b & 0x80000000u) ? ~b : (b | 0x80000000u);
}

__global__ void build_keys(const float* __restrict__ f,
                           unsigned long long* __restrict__ keys,
                           int sel, int nedge) {
    int i = blockIdx.x * blockDim.x + threadIdx.x;
    if (i >= NPAD) return;
    if (i >= nedge) { keys[i] = ~0ull; return; }  // pad sorts last
    int u, v;
    edge_uv(i, sel, u, v);
    __builtin_prefetch(f + u, 0, 0);              // global_prefetch_b8
    float wu = gval(f, u, sel), wv = gval(f, v, sel);
    float mx = fmaxf(wu, wv);
    // key = (ordered value) : (original index)  => stable ascending argsort
    keys[i] = ((unsigned long long)ordkey(mx) << 18) | (unsigned int)i;
}

// ------------------------------- sorting -----------------------------------
// Global compare-exchange pass (only for strides j >= SLOC).
__global__ void bitonic_step(unsigned long long* __restrict__ keys,
                             unsigned j, unsigned k) {
    unsigned i = blockIdx.x * blockDim.x + threadIdx.x;
    unsigned ixj = i ^ j;
    if (ixj > i) {
        unsigned long long a = keys[i], b = keys[ixj];
        bool swp = ((i & k) == 0) ? (a > b) : (a < b);
        if (swp) { keys[i] = b; keys[ixj] = a; }
    }
}

// Full bitonic sort of each SLOC-element chunk in LDS (stages k = 2..SLOC).
__global__ void bitonic_local_sort(unsigned long long* __restrict__ keys) {
    __shared__ unsigned long long sh[SLOC];
    unsigned base = blockIdx.x * SLOC;
    if (threadIdx.x < 32)                       // one wave issues the TDM DMA
        tdm_load_chunk(keys + base, (unsigned)(uintptr_t)&sh[0]);
    __syncthreads();
    for (unsigned t = threadIdx.x; t < SLOC; t += blockDim.x)
        sh[t] = keys[base + t];                 // authoritative overwrite
    __syncthreads();
    for (unsigned k = 2; k <= SLOC; k <<= 1) {
        for (unsigned j = k >> 1; j > 0; j >>= 1) {
            for (unsigned t = threadIdx.x; t < SLOC / 2; t += blockDim.x) {
                unsigned i = (t << 1) - (t & (j - 1));   // bit j of i is clear
                unsigned long long a = sh[i], b = sh[i + j];
                bool up = (((base + i) & k) == 0);
                if (up ? (a > b) : (a < b)) { sh[i] = b; sh[i + j] = a; }
            }
            __syncthreads();
        }
    }
    for (unsigned t = threadIdx.x; t < SLOC; t += blockDim.x)
        keys[base + t] = sh[t];
}

// Fused tail of one merge stage k: strides j = SLOC/2 .. 1 inside LDS.
__global__ void bitonic_fused(unsigned long long* __restrict__ keys, unsigned k) {
    __shared__ unsigned long long sh[SLOC];
    unsigned base = blockIdx.x * SLOC;
    if (threadIdx.x < 32)
        tdm_load_chunk(keys + base, (unsigned)(uintptr_t)&sh[0]);
    __syncthreads();
    for (unsigned t = threadIdx.x; t < SLOC; t += blockDim.x)
        sh[t] = keys[base + t];                 // authoritative overwrite
    __syncthreads();
    for (unsigned j = SLOC / 2; j > 0; j >>= 1) {
        for (unsigned t = threadIdx.x; t < SLOC / 2; t += blockDim.x) {
            unsigned i = (t << 1) - (t & (j - 1));
            unsigned long long a = sh[i], b = sh[i + j];
            bool up = (((base + i) & k) == 0);
            if (up ? (a > b) : (a < b)) { sh[i] = b; sh[i + j] = a; }
        }
        __syncthreads();
    }
    for (unsigned t = threadIdx.x; t < SLOC; t += blockDim.x)
        keys[base + t] = sh[t];
}

__global__ void fill_zero(float* __restrict__ p, int n) {
    int i = blockIdx.x * blockDim.x + threadIdx.x;
    if (i < n) p[i] = 0.0f;
}

// ----------------------------- union-find ----------------------------------
// Serial elder-rule union-find; parent[] lives in LDS (65537*4 B <= 320 KB WGP
// LDS). With parent[young]=old the surviving root IS the component birth
// vertex, so no birth array is needed and all comparisons (including <= / >=
// tie handling) exactly reproduce the reference.
__global__ void uf_serial(const float* __restrict__ f,
                          const unsigned long long* __restrict__ keys,
                          float* __restrict__ lengths,
                          float* __restrict__ accv,
                          int sel, int nedge) {
    extern __shared__ int parent[];
    int nv = sel ? (NVERT + 1) : NVERT;
    for (int x = threadIdx.x; x < nv; x += blockDim.x) parent[x] = x;
    __syncthreads();
    if (threadIdx.x != 0) return;

    float* lens = lengths + (sel ? NVERT : 0);
    int cnt = 0;
    float maxL = -INFINITY;

    for (int e = 0; e < nedge; ++e) {
        int i = (int)(keys[e] & 0x3FFFFull);
        int u, v;
        edge_uv(i, sel, u, v);

        int ra = u; while (parent[ra] != ra) ra = parent[ra];
        { int x = u; while (parent[x] != ra) { int nx = parent[x]; parent[x] = ra; x = nx; } }
        int rb = v; while (parent[rb] != rb) rb = parent[rb];
        { int x = v; while (parent[x] != rb) { int nx = parent[x]; parent[x] = rb; x = nx; } }
        if (ra == rb) continue;

        float wu = gval(f, u, sel), wv = gval(f, v, sel);
        int widx = (wu >= wv) ? u : v;                 // edge entry vertex (ties -> u)
        float gra = gval(f, ra, sel), grb = gval(f, rb, sel);
        int old_, young;
        if (gra <= grb) { old_ = ra; young = rb; } else { old_ = rb; young = ra; }

        float fb = f[young];   // young root == birth vertex, always a real vertex
        float fd = f[widx];    // widx is always a real vertex (virtual never wins max)
        float L = sel ? (fd - fb) : (fb - fd);
        lens[cnt++] = L;
        if (L > maxL) maxL = L;
        parent[young] = old_;
    }
    if (sel) accv[0] = maxL;   // longest H1 bar (first occurrence on ties)
}

// ---------------------------- WMMA reduction -------------------------------
// Sum 131072 floats via v_wmma_f32_16x16x32_f16. Each float -> hi+lo f16 pair
// (compensated split, ~2^-22 rel. error); B = ones so C rows accumulate in
// f32. 8 blocks x 256 threads = 64 waves, 8 WMMA iterations each, EXEC all-1s.
__global__ void wmma_reduce(const float* __restrict__ data,
                            float* __restrict__ partials) {
    int gtid = blockIdx.x * blockDim.x + threadIdx.x;
    int wave = gtid >> 5;
    int lane = gtid & 31;

    v8f c = {};
    v16h bmat;
#pragma unroll
    for (int q = 0; q < 16; ++q) bmat[q] = (_Float16)1.0f;

    const int nw = (8 * 256) >> 5;  // 64 waves; LEN_CAP / (nw*256) = 8 iterations
    for (int base = wave * 256; base < LEN_CAP; base += nw * 256) {
        const float* p = data + base + lane * 8;
        float4 x0 = ((const float4*)p)[0];
        float4 x1 = ((const float4*)p)[1];
        float xs[8] = {x0.x, x0.y, x0.z, x0.w, x1.x, x1.y, x1.z, x1.w};
        v16h a;
#pragma unroll
        for (int q = 0; q < 8; ++q) {
            float x = xs[q];
            _Float16 h = (_Float16)x;
            _Float16 l = (_Float16)(x - (float)h);
            a[2 * q]     = h;
            a[2 * q + 1] = l;
        }
        // (neg_a, A, neg_b, B, c_mod, C, reuse_a, reuse_b)
        c = __builtin_amdgcn_wmma_f32_16x16x32_f16(false, a, false, bmat,
                                                   (short)0, c, false, false);
    }
    // C[m][n] identical across n (B = ones): lanes 0..15 hold rows 0..7,
    // lanes 16..31 hold rows 8..15. Two partials per wave cover all rows.
    float s = c[0] + c[1] + c[2] + c[3] + c[4] + c[5] + c[6] + c[7];
    if (lane == 0)       partials[2 * wave]     = s;
    else if (lane == 16) partials[2 * wave + 1] = s;
}

// Deterministic finalize: fixed-order sum of 128 partials, minus longest H1 bar.
__global__ void finalize(const float* __restrict__ accv, float* __restrict__ out) {
    float s = 0.0f;
    for (int i = 0; i < 128; ++i) s += accv[2 + i];
    out[0] = s - accv[0];
}

// ------------------------------- driver ------------------------------------
extern "C" void kernel_launch(void* const* d_in, const int* in_sizes, int n_in,
                              void* d_out, int out_size, void* d_ws, size_t ws_size,
                              hipStream_t stream) {
    (void)in_sizes; (void)n_in; (void)out_size; (void)ws_size;
    const float* f = (const float*)d_in[0];
    float* out = (float*)d_out;

    char* ws = (char*)d_ws;
    unsigned long long* keys = (unsigned long long*)ws;            // 2 MB
    float* lengths = (float*)(ws + (size_t)NPAD * 8);              // 512 KB
    float* accv    = lengths + LEN_CAP;  // [0]=maxL, [2..129]=wave partials

    // Zero length slots + accumulator region (harness poisons ws with 0xAA).
    int nz = LEN_CAP + 256;
    fill_zero<<<(nz + 255) / 256, 256, 0, stream>>>(lengths, nz);

    for (int sel = 0; sel < 2; ++sel) {
        int ne = sel ? N_EDGE_B : N_EDGE_A;
        build_keys<<<NPAD / 256, 256, 0, stream>>>(f, keys, sel, ne);
        // Stages k = 2..SLOC entirely in LDS.
        bitonic_local_sort<<<NPAD / SLOC, 256, 0, stream>>>(keys);
        // Remaining stages: global passes while j >= SLOC, LDS-fused tail.
        for (unsigned k = SLOC * 2; k <= NPAD; k <<= 1) {
            for (unsigned j = k >> 1; j >= SLOC; j >>= 1)
                bitonic_step<<<NPAD / 256, 256, 0, stream>>>(keys, j, k);
            bitonic_fused<<<NPAD / SLOC, 256, 0, stream>>>(keys, k);
        }
        uf_serial<<<1, 256, (NVERT + 1) * sizeof(int), stream>>>(
            f, keys, lengths, accv, sel, ne);
    }

    wmma_reduce<<<8, 256, 0, stream>>>(lengths, accv + 2);
    finalize<<<1, 1, 0, stream>>>(accv, out);
}
